// MultiHeadAttention_86423331930678
// MI455X (gfx1250) — matrix operand order repaired and checked by reading, not compiled, not run on hardware
//
#include <hip/hip_runtime.h>
#include <hip/hip_bf16.h>

// ---------------- problem constants ----------------
#define BATCH   4
#define SEQ     2048
#define DMODEL  1024
#define NHEAD   16
#define DHEAD   64
#define MROWS   (BATCH * SEQ)        // 8192

typedef __bf16 v16bf __attribute__((ext_vector_type(16)));
typedef float  v8f   __attribute__((ext_vector_type(8)));
typedef int    v4i   __attribute__((vector_size(16)));

union ABu {            // one WMMA 16-bit operand fragment: 16 bf16 = 32 bytes
    v16bf    v;
    uint4    q[2];
    unsigned u[8];
};

__device__ __forceinline__ __bf16 f2bf(float f) {
    unsigned u = __builtin_bit_cast(unsigned, f);
    unsigned r = (u + 0x7FFFu + ((u >> 16) & 1u)) >> 16;   // round-nearest-even
    return __builtin_bit_cast(__bf16, (unsigned short)r);
}

// packed f32x2 -> bf16x2 (one v_cvt_pk_bf16_f32 when available)
__device__ __forceinline__ unsigned cvt2bf(float a, float b) {
#if defined(__HIP_DEVICE_COMPILE__) && __has_builtin(__builtin_amdgcn_cvt_pk_bf16_f32)
    typedef __bf16 v2bf __attribute__((ext_vector_type(2)));
    v2bf p = __builtin_amdgcn_cvt_pk_bf16_f32(a, b);
    return __builtin_bit_cast(unsigned, p);
#else
    unsigned lo = __builtin_bit_cast(unsigned short, f2bf(a));
    unsigned hi = __builtin_bit_cast(unsigned short, f2bf(b));
    return lo | (hi << 16);
#endif
}

// 16-byte global -> LDS copy; async (ASYNCcnt-tracked) path on gfx1250
__device__ __forceinline__ void cp_g2s_16(uint4* dst_lds, const uint4* src_glb) {
#if defined(__HIP_DEVICE_COMPILE__) && __has_builtin(__builtin_amdgcn_global_load_async_to_lds_b128)
    __builtin_amdgcn_global_load_async_to_lds_b128(
        (__attribute__((address_space(1))) v4i*)src_glb,
        (__attribute__((address_space(3))) v4i*)dst_lds, 0, 0);
#else
    *dst_lds = *src_glb;
#endif
}

__device__ __forceinline__ void cp_g2s_fence() {
#if defined(__HIP_DEVICE_COMPILE__) && __has_builtin(__builtin_amdgcn_global_load_async_to_lds_b128)
    asm volatile("s_wait_asynccnt 0" ::: "memory");
#endif
}

__device__ __forceinline__ v8f wmma_bf16(const v16bf& a, const v16bf& b, const v8f& c) {
    return __builtin_amdgcn_wmma_f32_16x16x32_bf16(
        /*neg_a=*/false, a, /*neg_b=*/false, b,
        /*c_mod=*/(short)0, c, /*reuse_a=*/false, /*reuse_b=*/false);
}

// ---------------- kernel 0: weight transpose f32[K][N] -> bf16 Wt[N][K] ----------------
__global__ __launch_bounds__(256)
void wtrans_kernel(const float* __restrict__ W, __bf16* __restrict__ Wt) {
    __shared__ float tile[32][33];
    int x = blockIdx.x * 32 + threadIdx.x;
    int y = blockIdx.y * 32 + threadIdx.y;
#pragma unroll
    for (int j = 0; j < 4; ++j)
        tile[threadIdx.y + j * 8][threadIdx.x] = W[(size_t)(y + j * 8) * DMODEL + x];
    __syncthreads();
    int xo = blockIdx.y * 32 + threadIdx.x;
    int yo = blockIdx.x * 32 + threadIdx.y;
#pragma unroll
    for (int j = 0; j < 4; ++j)
        Wt[(size_t)(yo + j * 8) * DMODEL + xo] = f2bf(tile[threadIdx.x][threadIdx.y + j * 8]);
}

// ---------------- kernel 1: projection GEMM  out = bf16(X @ W + b) ----------------
// X: f32 [8192][1024] row-major; Wt: bf16 [N][K]; out layout:
//   mode 0: q/k  bf16 [B][H][S][64]
//   mode 1: v^T  bf16 [B][H][64][S]
// Block = 8 waves; wave tile = 16(M) x 64(N); block tile = 128 x 64.
// B tile (64 cols x 32 k) double-buffered in LDS via async copies.
__global__ __launch_bounds__(256)
void proj_gemm_kernel(const float* __restrict__ X, const __bf16* __restrict__ Wt,
                      const float* __restrict__ bias, __bf16* __restrict__ out, int mode) {
    const int tid  = threadIdx.x;
    const int w    = tid >> 5;
    const int lane = tid & 31;
    const int lm   = lane & 15;
    const int lh   = lane >> 4;
    const int r0   = blockIdx.y * 128 + w * 16;
    const int n0   = blockIdx.x * 64;

    __shared__ uint4 Blds[2][64][4];                  // double-buffered [col][32 k bf16]

    const float* Arow = X + (size_t)(r0 + lm) * DMODEL;
    const int scol  = tid >> 2;
    const int spart = tid & 3;
    const __bf16* Bsrc = Wt + (size_t)(n0 + scol) * DMODEL + spart * 8;

    v8f acc[4];
#pragma unroll
    for (int t = 0; t < 4; ++t)
#pragma unroll
        for (int r = 0; r < 8; ++r) acc[t][r] = 0.0f;

    // prologue: stage first B tile
    cp_g2s_16(&Blds[0][scol][spart], (const uint4*)Bsrc);

    for (int kc = 0, i = 0; kc < DMODEL; kc += 32, ++i) {
        cp_g2s_fence();
        __syncthreads();                              // stage(i) visible to all waves
        if (kc + 32 < DMODEL)                         // overlap: stage(i+1) under compute
            cp_g2s_16(&Blds[(i + 1) & 1][scol][spart], (const uint4*)(Bsrc + kc + 32));
        const int cur = i & 1;

        // A fragment: elems 0..7 -> k = kc + lh*8 + 0..7 ; elems 8..15 -> +16
        ABu a;
        {
            const float4* p0 = (const float4*)(Arow + kc + lh * 8);
            const float4* p1 = (const float4*)(Arow + kc + 16 + lh * 8);
            float4 f0 = p0[0], f1 = p0[1], f2 = p1[0], f3 = p1[1];
            a.u[0] = cvt2bf(f0.x, f0.y); a.u[1] = cvt2bf(f0.z, f0.w);
            a.u[2] = cvt2bf(f1.x, f1.y); a.u[3] = cvt2bf(f1.z, f1.w);
            a.u[4] = cvt2bf(f2.x, f2.y); a.u[5] = cvt2bf(f2.z, f2.w);
            a.u[6] = cvt2bf(f3.x, f3.y); a.u[7] = cvt2bf(f3.z, f3.w);
        }

        // batch all B-fragment LDS loads, then issue the WMMA group
        ABu bfr[4];
#pragma unroll
        for (int t = 0; t < 4; ++t) {
            bfr[t].q[0] = Blds[cur][t * 16 + lm][lh];
            bfr[t].q[1] = Blds[cur][t * 16 + lm][2 + lh];
        }
#pragma unroll
        for (int t = 0; t < 4; ++t)
            acc[t] = wmma_bf16(a.v, bfr[t].v, acc[t]);
    }

#pragma unroll
    for (int t = 0; t < 4; ++t)
#pragma unroll
        for (int r = 0; r < 8; ++r) {
            int row = r0 + r + 8 * lh;                 // global M row = b*S + s
            int col = n0 + t * 16 + lm;                // d_model column = h*64 + d
            float val = acc[t][r] + bias[col];
            int b = row >> 11, s = row & (SEQ - 1);
            int h = col >> 6,  d = col & (DHEAD - 1);
            if (mode == 0)
                out[(((size_t)(b * NHEAD + h) * SEQ) + s) * DHEAD + d] = f2bf(val);
            else
                out[(((size_t)(b * NHEAD + h) * DHEAD) + d) * SEQ + s] = f2bf(val);
        }
}

// ---------------- kernel 2: fused flash attention ----------------
// q,k: bf16 [B][H][S][64]; vt: bf16 [B][H][64][S]; attn out: bf16 [B*S][1024]
// grid = (B*H, S/128); block = 8 waves; each wave owns 16 query rows.
// K/V chunks (32 keys) double-buffered in LDS via async copies.
__global__ __launch_bounds__(256)
void attn_kernel(const __bf16* __restrict__ q, const __bf16* __restrict__ k,
                 const __bf16* __restrict__ vt, __bf16* __restrict__ attn) {
    const float CS = 0.125f * 1.44269504f;             // (1/sqrt(64)) * log2(e)
    const int tid  = threadIdx.x;
    const int bh   = blockIdx.x;
    const int b    = bh >> 4;
    const int h    = bh & (NHEAD - 1);
    const int w    = tid >> 5;
    const int lane = tid & 31;
    const int lm   = lane & 15;
    const int lh   = lane >> 4;
    const int qbase = blockIdx.y * 128 + w * 16;

    const __bf16* qh  = q  + (size_t)bh * SEQ * DHEAD;
    const __bf16* kh  = k  + (size_t)bh * SEQ * DHEAD;
    const __bf16* vth = vt + (size_t)bh * DHEAD * SEQ;

    __shared__ uint4 Klds[2][32][8];                   // double-buffered [key][64 d bf16]
    __shared__ uint4 Vlds[2][64][4];                   // double-buffered [d][32 key bf16]
    __shared__ uint4 Plds[8][16][4];                   // per-wave 16x32 bf16 P

    const int kkey = tid >> 3, kpart = tid & 7;
    const int vd   = tid >> 2, vpart = tid & 3;
    const __bf16* Ksrc = kh + (size_t)kkey * DHEAD + kpart * 8;
    const __bf16* Vsrc = vth + (size_t)vd * SEQ + vpart * 8;

    // Q fragments held in registers for the whole key loop (d 0..31 and 32..63)
    ABu aq[2];
#pragma unroll
    for (int dh = 0; dh < 2; ++dh) {
        const __bf16* base = qh + (size_t)(qbase + lm) * DHEAD + dh * 32;
        aq[dh].q[0] = *(const uint4*)(base + lh * 8);
        aq[dh].q[1] = *(const uint4*)(base + 16 + lh * 8);
    }

    v8f o[4];
    float mr[8], lr[8];
#pragma unroll
    for (int t = 0; t < 4; ++t)
#pragma unroll
        for (int r = 0; r < 8; ++r) o[t][r] = 0.0f;
#pragma unroll
    for (int r = 0; r < 8; ++r) { mr[r] = -1e30f; lr[r] = 0.0f; }

    // prologue: stage first K/V chunk
    cp_g2s_16(&Klds[0][kkey][kpart], (const uint4*)Ksrc);
    cp_g2s_16(&Vlds[0][vd][vpart],   (const uint4*)Vsrc);

    for (int kc = 0, i = 0; kc < SEQ; kc += 32, ++i) {
        cp_g2s_fence();
        __syncthreads();                              // stage(i) visible; buffers reusable
        if (kc + 32 < SEQ) {                          // overlap: stage(i+1) under compute
            cp_g2s_16(&Klds[(i + 1) & 1][kkey][kpart],
                      (const uint4*)(Ksrc + (size_t)(kc + 32) * DHEAD));
            cp_g2s_16(&Vlds[(i + 1) & 1][vd][vpart],
                      (const uint4*)(Vsrc + kc + 32));
        }
        const int cur = i & 1;

        // scores = Q @ K^T : batch all K-fragment loads, then 4 WMMAs
        ABu bk[2][2];                                  // [dh][key-tile]
#pragma unroll
        for (int dh = 0; dh < 2; ++dh)
#pragma unroll
            for (int t = 0; t < 2; ++t) {
                bk[dh][t].q[0] = Klds[cur][t * 16 + lm][dh * 4 + lh];
                bk[dh][t].q[1] = Klds[cur][t * 16 + lm][dh * 4 + 2 + lh];
            }
        v8f s0, s1;
#pragma unroll
        for (int r = 0; r < 8; ++r) { s0[r] = 0.0f; s1[r] = 0.0f; }
        s0 = wmma_bf16(aq[0].v, bk[0][0].v, s0);
        s1 = wmma_bf16(aq[0].v, bk[0][1].v, s1);
        s0 = wmma_bf16(aq[1].v, bk[1][0].v, s0);
        s1 = wmma_bf16(aq[1].v, bk[1][1].v, s1);

        // online softmax per accumulator register r (row = r + 8*lh)
#pragma unroll
        for (int r = 0; r < 8; ++r) {
            float mx = fmaxf(s0[r], s1[r]);
            mx = fmaxf(mx, __shfl_xor(mx, 1));
            mx = fmaxf(mx, __shfl_xor(mx, 2));
            mx = fmaxf(mx, __shfl_xor(mx, 4));
            mx = fmaxf(mx, __shfl_xor(mx, 8));
            float mnew  = fmaxf(mr[r], mx);
            float alpha = exp2f((mr[r] - mnew) * CS);
            mr[r] = mnew;
            float p0 = exp2f((s0[r] - mnew) * CS);
            float p1 = exp2f((s1[r] - mnew) * CS);
            float ps = p0 + p1;
            ps += __shfl_xor(ps, 1);
            ps += __shfl_xor(ps, 2);
            ps += __shfl_xor(ps, 4);
            ps += __shfl_xor(ps, 8);
            lr[r] = lr[r] * alpha + ps;
            o[0][r] *= alpha; o[1][r] *= alpha; o[2][r] *= alpha; o[3][r] *= alpha;
            __bf16* prow = (__bf16*)&Plds[w][r + 8 * lh][0];
            prow[lm]      = f2bf(p0);
            prow[lm + 16] = f2bf(p1);
        }
        __syncthreads();                              // P staged (per-wave region)

        // O += P @ V : batch P/V fragment loads, then 4 WMMAs
        ABu pa;
        pa.q[0] = Plds[w][lm][lh];
        pa.q[1] = Plds[w][lm][2 + lh];
        ABu vb[4];
#pragma unroll
        for (int t = 0; t < 4; ++t) {
            vb[t].q[0] = Vlds[cur][t * 16 + lm][lh];
            vb[t].q[1] = Vlds[cur][t * 16 + lm][2 + lh];
        }
#pragma unroll
        for (int t = 0; t < 4; ++t)
            o[t] = wmma_bf16(pa.v, vb[t].v, o[t]);
    }

    // normalize + store to attn [B*S][1024] (dmodel col = h*64 + d)
#pragma unroll
    for (int r = 0; r < 8; ++r) {
        float inv = 1.0f / lr[r];
        int sq = qbase + r + 8 * lh;
        size_t rowoff = ((size_t)b * SEQ + sq) * DMODEL + h * DHEAD;
#pragma unroll
        for (int t = 0; t < 4; ++t)
            attn[rowoff + t * 16 + lm] = f2bf(o[t][r] * inv);
    }
}

// ---------------- kernel 3: output GEMM  d_out = f32(attn_bf16 @ W_O + b_O) ----------------
__global__ __launch_bounds__(256)
void out_gemm_kernel(const __bf16* __restrict__ A, const __bf16* __restrict__ Wt,
                     const float* __restrict__ bias, float* __restrict__ out) {
    const int tid  = threadIdx.x;
    const int w    = tid >> 5;
    const int lane = tid & 31;
    const int lm   = lane & 15;
    const int lh   = lane >> 4;
    const int r0   = blockIdx.y * 128 + w * 16;
    const int n0   = blockIdx.x * 64;

    __shared__ uint4 Blds[2][64][4];

    const __bf16* Arow = A + (size_t)(r0 + lm) * DMODEL;
    const int scol  = tid >> 2;
    const int spart = tid & 3;
    const __bf16* Bsrc = Wt + (size_t)(n0 + scol) * DMODEL + spart * 8;

    v8f acc[4];
#pragma unroll
    for (int t = 0; t < 4; ++t)
#pragma unroll
        for (int r = 0; r < 8; ++r) acc[t][r] = 0.0f;

    cp_g2s_16(&Blds[0][scol][spart], (const uint4*)Bsrc);

    for (int kc = 0, i = 0; kc < DMODEL; kc += 32, ++i) {
        cp_g2s_fence();
        __syncthreads();
        if (kc + 32 < DMODEL)
            cp_g2s_16(&Blds[(i + 1) & 1][scol][spart], (const uint4*)(Bsrc + kc + 32));
        const int cur = i & 1;

        ABu a;
        a.q[0] = *(const uint4*)(Arow + kc + lh * 8);
        a.q[1] = *(const uint4*)(Arow + kc + 16 + lh * 8);

        ABu bfr[4];
#pragma unroll
        for (int t = 0; t < 4; ++t) {
            bfr[t].q[0] = Blds[cur][t * 16 + lm][lh];
            bfr[t].q[1] = Blds[cur][t * 16 + lm][2 + lh];
        }
#pragma unroll
        for (int t = 0; t < 4; ++t)
            acc[t] = wmma_bf16(a.v, bfr[t].v, acc[t]);
    }

#pragma unroll
    for (int t = 0; t < 4; ++t)
#pragma unroll
        for (int r = 0; r < 8; ++r) {
            int row = r0 + r + 8 * lh;
            int col = n0 + t * 16 + lm;
            out[(size_t)row * DMODEL + col] = acc[t][r] + bias[col];
        }
}

// ---------------- host-side launcher ----------------
extern "C" void kernel_launch(void* const* d_in, const int* in_sizes, int n_in,
                              void* d_out, int out_size, void* d_ws, size_t ws_size,
                              hipStream_t stream) {
    (void)in_sizes; (void)n_in; (void)out_size; (void)ws_size;
    const float* Qin = (const float*)d_in[0];
    const float* Kin = (const float*)d_in[1];
    const float* Vin = (const float*)d_in[2];
    const float* WQ  = (const float*)d_in[3];
    const float* bQ  = (const float*)d_in[4];
    const float* WK  = (const float*)d_in[5];
    const float* bK  = (const float*)d_in[6];
    const float* WV  = (const float*)d_in[7];
    const float* bV  = (const float*)d_in[8];
    const float* WO  = (const float*)d_in[9];
    const float* bO  = (const float*)d_in[10];

    const size_t WT_ELEMS  = (size_t)DMODEL * DMODEL;             // 1M bf16
    const size_t QKV_ELEMS = (size_t)BATCH * NHEAD * SEQ * DHEAD; // 8M bf16
    __bf16* WtQ   = (__bf16*)d_ws;
    __bf16* WtK   = WtQ + WT_ELEMS;
    __bf16* WtV   = WtK + WT_ELEMS;
    __bf16* WtO   = WtV + WT_ELEMS;
    __bf16* qb    = WtO + WT_ELEMS;
    __bf16* kb    = qb + QKV_ELEMS;
    __bf16* vtb   = kb + QKV_ELEMS;
    __bf16* attnb = vtb + QKV_ELEMS;

    dim3 tb(32, 8), tg(DMODEL / 32, DMODEL / 32);
    wtrans_kernel<<<tg, tb, 0, stream>>>(WQ, WtQ);
    wtrans_kernel<<<tg, tb, 0, stream>>>(WK, WtK);
    wtrans_kernel<<<tg, tb, 0, stream>>>(WV, WtV);
    wtrans_kernel<<<tg, tb, 0, stream>>>(WO, WtO);

    dim3 gg(DMODEL / 64, MROWS / 128);  // (16, 64)
    proj_gemm_kernel<<<gg, 256, 0, stream>>>(Qin, WtQ, bQ, qb, 0);
    proj_gemm_kernel<<<gg, 256, 0, stream>>>(Kin, WtK, bK, kb, 0);
    proj_gemm_kernel<<<gg, 256, 0, stream>>>(Vin, WtV, bV, vtb, 1);

    attn_kernel<<<dim3(BATCH * NHEAD, SEQ / 128), 256, 0, stream>>>(qb, kb, vtb, attnb);

    out_gemm_kernel<<<gg, 256, 0, stream>>>(attnb, WtO, bO, (float*)d_out);
}
